// Router_54391465837033
// MI455X (gfx1250) — matrix operand order repaired and checked by reading, not compiled, not run on hardware
//
#include <hip/hip_runtime.h>

// ---------------------------------------------------------------------------
// MoE router for MI455X (gfx1250, wave32).
// GEMM (32768 x 4096) @ (4096 x 64)^T as 3xBF16 split-precision WMMA
// (v_wmma_f32_16x16x32_bf16)  -> ~fp32 routing accuracy, HBM-bound
// (~512MB / 23.3 TB/s ~= 22us).
// W is prepacked once into d_ws as bf16 hi/lo planes (chunk-major), and the
// main kernel stages W chunks with double-buffered ASYNC copies
// (global_load_async_to_lds_b128, ASYNCcnt) instead of re-converting f32->bf16
// in every block every chunk.
// ---------------------------------------------------------------------------

typedef __attribute__((ext_vector_type(16))) __bf16 v16bf;
typedef __attribute__((ext_vector_type(8)))  float  v8f;
typedef int v4i __attribute__((vector_size(16)));   // matches builtin param type

#define HID      4096
#define NEXP     64
#define KC       128            // K per staged chunk
#define NCHUNK   (HID / KC)     // 32
#define W_ROW_DW 68             // LDS row stride (dwords) for bf16 W rows (272B, 16B-aligned)
#define LOGROW   65             // logits row stride (floats) to dodge bank conflicts
#define BUFSTRIDE (2 * NEXP * W_ROW_DW)    // dwords per double-buffer slot (hi+lo)
#define WS_PLANE_DW (NCHUNK * NEXP * 64)   // dwords per prepacked plane (hi or lo)

#define AS1 __attribute__((address_space(1)))
#define AS3 __attribute__((address_space(3)))

// Truncation split of two f32 into packed bf16-hi dword + bf16-lo dword.
__device__ __forceinline__ void split2(float a, float b,
                                       unsigned int& hi, unsigned int& lo) {
    unsigned int ua = __float_as_uint(a), ub = __float_as_uint(b);
    hi = (ua >> 16) | (ub & 0xffff0000u);
    float ha = __uint_as_float(ua & 0xffff0000u);
    float hb = __uint_as_float(ub & 0xffff0000u);
    float la = a - ha, lb = b - hb;
    lo = (__float_as_uint(la) >> 16) | (__float_as_uint(lb) & 0xffff0000u);
}

union ABits { unsigned int u[8]; uint4 q[2]; v16bf v; };

// ---------------------------------------------------------------------------
// Pass 1: prepack router_w (64 x 4096 f32) into bf16 hi/lo planes, chunk-major:
// plane[((c*64 + e)*64) + k/2] = pack(W[e][c*128+k], W[e][c*128+k+1]).
// Each K-chunk is a contiguous 64*256B block per plane -> async-copy friendly.
// ---------------------------------------------------------------------------
__global__ __launch_bounds__(256)
void prepack_w_kernel(const float* __restrict__ W,
                      unsigned int* __restrict__ wsHi,
                      unsigned int* __restrict__ wsLo) {
    int g = blockIdx.x * 256 + threadIdx.x;     // 0..65535, 4 floats each
    int e = g >> 10;                            // 1024 threads per expert row
    int k = (g & 1023) * 4;                     // column within expert row
    const float4 f = *(const float4*)(W + (size_t)e * HID + k);
    unsigned int h0, h1, l0, l1;
    split2(f.x, f.y, h0, l0);
    split2(f.z, f.w, h1, l1);
    int c   = k >> 7;
    int kin = k & (KC - 1);
    unsigned int idx = ((unsigned)(c * NEXP + e)) * 64 + (kin >> 1);
    wsHi[idx] = h0; wsHi[idx + 1] = h1;
    wsLo[idx] = l0; wsLo[idx + 1] = l1;
}

// Issue this thread's share (4+4 b128 async transfers) of one W-chunk copy.
__device__ __forceinline__ void issue_w_copy(const unsigned int* wsHi,
                                             const unsigned int* wsLo,
                                             unsigned int* bufHi,
                                             unsigned int* bufLo,
                                             int c, int e, int sub) {
    AS1 v4i* gh = (AS1 v4i*)(wsHi + ((size_t)(c * NEXP + e)) * 64 + sub * 16);
    AS1 v4i* gl = (AS1 v4i*)(wsLo + ((size_t)(c * NEXP + e)) * 64 + sub * 16);
    AS3 v4i* dh = (AS3 v4i*)(bufHi + e * W_ROW_DW + sub * 16);
    AS3 v4i* dl = (AS3 v4i*)(bufLo + e * W_ROW_DW + sub * 16);
    __builtin_amdgcn_global_load_async_to_lds_b128(gh, dh,  0, 0);
    __builtin_amdgcn_global_load_async_to_lds_b128(gh, dh, 16, 0);
    __builtin_amdgcn_global_load_async_to_lds_b128(gh, dh, 32, 0);
    __builtin_amdgcn_global_load_async_to_lds_b128(gh, dh, 48, 0);
    __builtin_amdgcn_global_load_async_to_lds_b128(gl, dl,  0, 0);
    __builtin_amdgcn_global_load_async_to_lds_b128(gl, dl, 16, 0);
    __builtin_amdgcn_global_load_async_to_lds_b128(gl, dl, 32, 0);
    __builtin_amdgcn_global_load_async_to_lds_b128(gl, dl, 48, 0);
}

// ---------------------------------------------------------------------------
// Pass 2: main GEMM + softmax + top2 with async double-buffered W staging.
// Block = 256 threads (8 waves); wave owns 16 tokens x 64 experts.
// ---------------------------------------------------------------------------
__global__ __launch_bounds__(256)
void router_topk_wmma_async(const float* __restrict__ X,
                            const unsigned int* __restrict__ wsHi,
                            const unsigned int* __restrict__ wsLo,
                            float* __restrict__ out,
                            int n_tokens) {
    // Two W buffers (hi+lo each): 4 * 64 * 68 dwords = 69632 B.
    // Logits (8*16*65 floats = 33280 B) alias buffer 0 after the K loop.
    __shared__ unsigned int lds[2 * BUFSTRIDE];

    const int tid  = threadIdx.x;
    const int lane = tid & 31;
    const int wave = tid >> 5;
    const int hi16 = lane >> 4;
    const int l15  = lane & 15;
    const int erow = tid >> 2;     // expert row this thread copies
    const int esub = tid & 3;      // 16-dword sub-slice within the row

    const int wave_token0 = blockIdx.x * 128 + wave * 16;
    const float* xrow = X + (size_t)(wave_token0 + l15) * HID;

    v8f acc[4] = {};

    // Prologue: kick off chunk 0 into buffer 0.
    issue_w_copy(wsHi, wsLo, lds, lds + NEXP * W_ROW_DW, 0, erow, esub);

    for (int c = 0; c < NCHUNK; ++c) {
        asm volatile("s_wait_asynccnt 0x0" ::: "memory");  // my copies landed
        __syncthreads();  // everyone's copies landed; prior-buffer reads done

        if (c + 1 < NCHUNK) {
            unsigned int* nb = lds + ((c + 1) & 1) * BUFSTRIDE;
            issue_w_copy(wsHi, wsLo, nb, nb + NEXP * W_ROW_DW,
                         c + 1, erow, esub);
        }

        const unsigned int* bh_base = lds + (c & 1) * BUFSTRIDE;
        const unsigned int* bl_base = bh_base + NEXP * W_ROW_DW;

#pragma unroll
        for (int j = 0; j < 4; ++j) {
            // A 16x32, contiguous k order (same as B): lane holds K = {16*hi16..+15}
            // of its token row, 2 per VGPR.
            const float* xp = xrow + c * KC + j * 32 + hi16 * 16;
            float4 f0 = *(const float4*)(xp);
            float4 f1 = *(const float4*)(xp + 4);
            float4 f2 = *(const float4*)(xp + 8);
            float4 f3 = *(const float4*)(xp + 12);

            ABits ahi, alo;
            split2(f0.x, f0.y, ahi.u[0], alo.u[0]);
            split2(f0.z, f0.w, ahi.u[1], alo.u[1]);
            split2(f1.x, f1.y, ahi.u[2], alo.u[2]);
            split2(f1.z, f1.w, ahi.u[3], alo.u[3]);
            split2(f2.x, f2.y, ahi.u[4], alo.u[4]);
            split2(f2.z, f2.w, ahi.u[5], alo.u[5]);
            split2(f3.x, f3.y, ahi.u[6], alo.u[6]);
            split2(f3.z, f3.w, ahi.u[7], alo.u[7]);

#pragma unroll
            for (int nt = 0; nt < 4; ++nt) {
                // B 32x16, contiguous k order (same as A): lane = expert column; first 8 VGPRs
                // hold K 0..15 (lanes 0-15) / 16..31 (lanes 16-31).
                const unsigned int* bh =
                    bh_base + (nt * 16 + l15) * W_ROW_DW + j * 16 + hi16 * 8;
                const unsigned int* bl =
                    bl_base + (nt * 16 + l15) * W_ROW_DW + j * 16 + hi16 * 8;
                ABits bhi, blo;
                bhi.q[0] = *(const uint4*)(bh);
                bhi.q[1] = *(const uint4*)(bh + 4);
                blo.q[0] = *(const uint4*)(bl);
                blo.q[1] = *(const uint4*)(bl + 4);

                acc[nt] = __builtin_amdgcn_wmma_f32_16x16x32_bf16(
                    false, ahi.v, false, bhi.v, (short)0, acc[nt], false, false);
                acc[nt] = __builtin_amdgcn_wmma_f32_16x16x32_bf16(
                    false, ahi.v, false, blo.v, (short)0, acc[nt], false, false);
                acc[nt] = __builtin_amdgcn_wmma_f32_16x16x32_bf16(
                    false, alo.v, false, bhi.v, (short)0, acc[nt], false, false);
            }
        }
    }

    // ---- epilogue: dump logits to LDS (aliases buffer 0), softmax + top2 ---
    __syncthreads();
    float* lds_logits = (float*)lds;
    {
        float* lrow = lds_logits + wave * (16 * LOGROW);
#pragma unroll
        for (int v = 0; v < 8; ++v) {
            int tl = v + 8 * hi16;  // C/D: VGPR v -> M=v / M=v+8 per half-wave
            lrow[tl * LOGROW +  0 + l15] = acc[0][v];
            lrow[tl * LOGROW + 16 + l15] = acc[1][v];
            lrow[tl * LOGROW + 32 + l15] = acc[2][v];
            lrow[tl * LOGROW + 48 + l15] = acc[3][v];
        }
    }
    __syncthreads();

    if (lane < 16) {
        const float* xr = lds_logits + wave * (16 * LOGROW) + lane * LOGROW;
        float m = -INFINITY;
#pragma unroll 8
        for (int e = 0; e < NEXP; ++e) m = fmaxf(m, xr[e]);
        float sum = 0.f;
        float v1 = -INFINITY, v2 = -INFINITY;
        int   i1 = 0, i2 = 0;
        for (int e = 0; e < NEXP; ++e) {
            float x = xr[e];
            sum += __expf(x - m);
            if (x > v1)      { v2 = v1; i2 = i1; v1 = x; i1 = e; }
            else if (x > v2) { v2 = x;  i2 = e; }
        }
        float inv = 1.f / sum;
        float p1 = __expf(v1 - m) * inv;
        float p2 = __expf(v2 - m) * inv;
        // reference renormalizes with softmax over the two *probabilities*
        float s1 = 1.f / (1.f + __expf(p2 - p1));
        float s2 = 1.f - s1;

        int token = wave_token0 + lane;
        out[2 * token]     = s1;
        out[2 * token + 1] = s2;
        float* oidx = out + 2 * (size_t)n_tokens;
        oidx[2 * token]     = (float)i1;
        oidx[2 * token + 1] = (float)i2;
    }
}

// ---------------------------------------------------------------------------
// Fallback (no workspace): self-contained version converting W in-kernel.
// ---------------------------------------------------------------------------
__global__ __launch_bounds__(256)
void router_topk_wmma_fallback(const float* __restrict__ X,
                               const float* __restrict__ W,
                               float* __restrict__ out,
                               int n_tokens) {
    __shared__ unsigned int lds[2 * NEXP * W_ROW_DW];
    unsigned int* lds_whi = lds;
    unsigned int* lds_wlo = lds + NEXP * W_ROW_DW;

    const int tid  = threadIdx.x;
    const int lane = tid & 31;
    const int wave = tid >> 5;
    const int hi16 = lane >> 4;
    const int l15  = lane & 15;

    const int wave_token0 = blockIdx.x * 128 + wave * 16;
    const float* xrow = X + (size_t)(wave_token0 + l15) * HID;

    v8f acc[4] = {};

    for (int c = 0; c < NCHUNK; ++c) {
        __syncthreads();
        {
            const int e   = tid >> 2;
            const int sub = tid & 3;
            const float4* gw = (const float4*)(W + (size_t)e * HID + c * KC + sub * 32);
            unsigned int* whi = lds_whi + e * W_ROW_DW + sub * 16;
            unsigned int* wlo = lds_wlo + e * W_ROW_DW + sub * 16;
#pragma unroll
            for (int j = 0; j < 8; ++j) {
                float4 f = gw[j];
                unsigned int h0, h1, l0, l1;
                split2(f.x, f.y, h0, l0);
                split2(f.z, f.w, h1, l1);
                whi[2 * j] = h0; whi[2 * j + 1] = h1;
                wlo[2 * j] = l0; wlo[2 * j + 1] = l1;
            }
        }
        __syncthreads();

#pragma unroll
        for (int j = 0; j < 4; ++j) {
            const float* xp = xrow + c * KC + j * 32 + hi16 * 16;
            float4 f0 = *(const float4*)(xp);
            float4 f1 = *(const float4*)(xp + 4);
            float4 f2 = *(const float4*)(xp + 8);
            float4 f3 = *(const float4*)(xp + 12);

            ABits ahi, alo;
            split2(f0.x, f0.y, ahi.u[0], alo.u[0]);
            split2(f0.z, f0.w, ahi.u[1], alo.u[1]);
            split2(f1.x, f1.y, ahi.u[2], alo.u[2]);
            split2(f1.z, f1.w, ahi.u[3], alo.u[3]);
            split2(f2.x, f2.y, ahi.u[4], alo.u[4]);
            split2(f2.z, f2.w, ahi.u[5], alo.u[5]);
            split2(f3.x, f3.y, ahi.u[6], alo.u[6]);
            split2(f3.z, f3.w, ahi.u[7], alo.u[7]);

#pragma unroll
            for (int nt = 0; nt < 4; ++nt) {
                const unsigned int* bh =
                    lds_whi + (nt * 16 + l15) * W_ROW_DW + j * 16 + hi16 * 8;
                const unsigned int* bl =
                    lds_wlo + (nt * 16 + l15) * W_ROW_DW + j * 16 + hi16 * 8;
                ABits bhi, blo;
                bhi.q[0] = *(const uint4*)(bh);
                bhi.q[1] = *(const uint4*)(bh + 4);
                blo.q[0] = *(const uint4*)(bl);
                blo.q[1] = *(const uint4*)(bl + 4);

                acc[nt] = __builtin_amdgcn_wmma_f32_16x16x32_bf16(
                    false, ahi.v, false, bhi.v, (short)0, acc[nt], false, false);
                acc[nt] = __builtin_amdgcn_wmma_f32_16x16x32_bf16(
                    false, ahi.v, false, blo.v, (short)0, acc[nt], false, false);
                acc[nt] = __builtin_amdgcn_wmma_f32_16x16x32_bf16(
                    false, alo.v, false, bhi.v, (short)0, acc[nt], false, false);
            }
        }
    }

    __syncthreads();
    float* lds_logits = (float*)lds;
    {
        float* lrow = lds_logits + wave * (16 * LOGROW);
#pragma unroll
        for (int v = 0; v < 8; ++v) {
            int tl = v + 8 * hi16;
            lrow[tl * LOGROW +  0 + l15] = acc[0][v];
            lrow[tl * LOGROW + 16 + l15] = acc[1][v];
            lrow[tl * LOGROW + 32 + l15] = acc[2][v];
            lrow[tl * LOGROW + 48 + l15] = acc[3][v];
        }
    }
    __syncthreads();

    if (lane < 16) {
        const float* xr = lds_logits + wave * (16 * LOGROW) + lane * LOGROW;
        float m = -INFINITY;
#pragma unroll 8
        for (int e = 0; e < NEXP; ++e) m = fmaxf(m, xr[e]);
        float sum = 0.f;
        float v1 = -INFINITY, v2 = -INFINITY;
        int   i1 = 0, i2 = 0;
        for (int e = 0; e < NEXP; ++e) {
            float x = xr[e];
            sum += __expf(x - m);
            if (x > v1)      { v2 = v1; i2 = i1; v1 = x; i1 = e; }
            else if (x > v2) { v2 = x;  i2 = e; }
        }
        float inv = 1.f / sum;
        float p1 = __expf(v1 - m) * inv;
        float p2 = __expf(v2 - m) * inv;
        float s1 = 1.f / (1.f + __expf(p2 - p1));
        float s2 = 1.f - s1;

        int token = wave_token0 + lane;
        out[2 * token]     = s1;
        out[2 * token + 1] = s2;
        float* oidx = out + 2 * (size_t)n_tokens;
        oidx[2 * token]     = (float)i1;
        oidx[2 * token + 1] = (float)i2;
    }
}

extern "C" void kernel_launch(void* const* d_in, const int* in_sizes, int n_in,
                              void* d_out, int out_size, void* d_ws, size_t ws_size,
                              hipStream_t stream) {
    const float* X = (const float*)d_in[0];   // hidden_states (4,8192,4096) f32
    const float* W = (const float*)d_in[1];   // router_w (64,4096) f32
    float* out = (float*)d_out;               // scores then indices, flat
    int n_tokens = in_sizes[0] / HID;         // 32768
    int blocks = n_tokens / 128;              // 256 blocks of 8 waves

    const size_t ws_needed = 2 * (size_t)WS_PLANE_DW * sizeof(unsigned int); // 1 MB
    if (ws_size >= ws_needed) {
        unsigned int* wsHi = (unsigned int*)d_ws;
        unsigned int* wsLo = wsHi + WS_PLANE_DW;
        prepack_w_kernel<<<dim3(256), dim3(256), 0, stream>>>(W, wsHi, wsLo);
        router_topk_wmma_async<<<dim3(blocks), dim3(256), 0, stream>>>(
            X, wsHi, wsLo, out, n_tokens);
    } else {
        router_topk_wmma_fallback<<<dim3(blocks), dim3(256), 0, stream>>>(
            X, W, out, n_tokens);
    }
}
